// LoretzFusion2_86337432584667
// MI455X (gfx1250) — matrix operand order repaired and checked by reading, not compiled, not run on hardware
//
#include <hip/hip_runtime.h>
#include <hip/hip_bf16.h>
#include <math.h>

#define N_ENT 10000
#define PADN  10048            // N_ENT rounded up to 64
#define MM_DIM 768
#define ST_DIM 256
#define COMPS 5
#define INV_TAU 10.0f
#define TOPK 6
#define BN_EPS 1e-5f
#define LDB 264                // LDS tile row stride (f16): 528B = 132 dwords -> bank rotate

typedef __attribute__((ext_vector_type(16))) _Float16 v16h;
typedef __attribute__((ext_vector_type(8)))  _Float16 v8h;
typedef __attribute__((ext_vector_type(8)))  float    v8f;

// ---------------------------------------------------------------------------
// WMMA fragment loaders (wave32, V_WMMA_F32_16X16X32_F16 layouts per ISA 7.12.2)
//  A 16x32 f16: lanes 0-15 -> M=lane,   K = {0..7, 16..23}
//               lanes16-31 -> M=lane-16,K = {8..15,24..31}
//  B 32x16 f16: lanes 0-15 -> N=lane,   K = {0..7, 16..23}
//               lanes16-31 -> N=lane-16,K = {8..15,24..31}  (B[k][n] = Bt[n][k], Bt row-major)
//  C 16x16 f32: vgpr v, lanes 0-15 -> (M=v,  N=lane); lanes16-31 -> (M=v+8, N=lane-16)
// All loaders are UNGUARDED: buffers are padded so every row is readable.
// ---------------------------------------------------------------------------

__device__ __forceinline__ v16h frag_a(const _Float16* __restrict__ A, int row0,
                                       int k0, int lda) {
  const int lane = threadIdx.x & 31;
  const int hl   = lane >> 4;
  const _Float16* p = A + (size_t)(row0 + (lane & 15)) * lda + k0;
  v8h lo = *(const v8h*)(p + hl * 8);
  v8h hi = *(const v8h*)(p + 16 + hl * 8);
  v16h f;
#pragma unroll
  for (int i = 0; i < 8; ++i) { f[i] = lo[i]; f[8 + i] = hi[i]; }
  return f;
}

__device__ __forceinline__ v16h frag_b(const _Float16* __restrict__ Bt, int n0,
                                       int k0, int ldb) {
  const int lane = threadIdx.x & 31;
  const int hl   = lane >> 4;
  const _Float16* p = Bt + (size_t)(n0 + (lane & 15)) * ldb + k0;
  v8h lo = *(const v8h*)(p + hl * 8);
  v8h hi = *(const v8h*)(p + 16 + hl * 8);
  v16h f;
#pragma unroll
  for (int i = 0; i < 8; ++i) { f[i] = lo[i]; f[8 + i] = hi[i]; }
  return f;
}

__device__ __forceinline__ v8f wmma_f16(v16h a, v16h b, v8f c) {
  return __builtin_amdgcn_wmma_f32_16x16x32_f16(false, a, false, b, (short)0, c,
                                                false, false);
}

// Stage 64 contiguous rows (256 f16 each) into an LDS tile with row stride LDB.
__device__ __forceinline__ void stage_contig(_Float16* dst, const _Float16* __restrict__ src) {
#pragma unroll
  for (int cnk = threadIdx.x; cnk < 2048; cnk += 256) {
    int row = cnk >> 5, col = (cnk & 31) << 3;
    *(v8h*)(dst + row * LDB + col) = *(const v8h*)(src + row * 256 + col);
  }
}

// Stage 64 gathered rows into an LDS tile.
__device__ __forceinline__ void stage_gather(_Float16* dst, const _Float16* __restrict__ base,
                                             const int* rows) {
#pragma unroll
  for (int cnk = threadIdx.x; cnk < 2048; cnk += 256) {
    int row = cnk >> 5, col = (cnk & 31) << 3;
    *(v8h*)(dst + row * LDB + col) = *(const v8h*)(base + (size_t)rows[row] * 256 + col);
  }
}

// ---------------------------------------------------------------------------
// Elementwise helpers
// ---------------------------------------------------------------------------
__global__ __launch_bounds__(256)
void f32_to_f16_kernel(const float* __restrict__ in, _Float16* __restrict__ out, int n) {
  int i = blockIdx.x * 256 + threadIdx.x;
  if (i < n) out[i] = (_Float16)in[i];
}

__global__ __launch_bounds__(256)
void zero_f16_kernel(_Float16* __restrict__ p, int n) {
  int i = blockIdx.x * 256 + threadIdx.x;
  if (i < n) p[i] = (_Float16)0.f;
}

// ---------------------------------------------------------------------------
// Encoder GEMM: out[e] = A[e] (MxK, padded rows) @ W[e]^T (KxN=256) + bias[e]
// do_gelu=1: exact GELU, write f16; else write f32.
// 64x64 tile / workgroup, 8 waves: col tile (w&3), row tiles {w>>2, w>>2+2}
// ---------------------------------------------------------------------------
__global__ __launch_bounds__(256)
void enc_gemm_kernel(const _Float16* __restrict__ A, size_t Astride, int lda,
                     const _Float16* __restrict__ W, size_t Wstride, int ldw,
                     const float* __restrict__ bias,
                     _Float16* __restrict__ outH, float* __restrict__ outF,
                     size_t OstrideE, int OstrideRow, int M, int K, int do_gelu) {
  const int e     = blockIdx.z;
  const int rbase = blockIdx.x * 64;
  const int nbase = blockIdx.y * 64;
  const _Float16* Ae = A + (size_t)e * Astride;
  const _Float16* We = W + (size_t)e * Wstride;
  const float*    be = bias + (size_t)e * ST_DIM;
  const int w    = threadIdx.x >> 5;
  const int lane = threadIdx.x & 31;
  const int nt   = w & 3;
  const int mt0  = w >> 2;

  v8f c0 = {}; v8f c1 = {};
  for (int k0 = 0; k0 < K; k0 += 32) {
    v16h b  = frag_b(We, nbase + nt * 16, k0, ldw);
    v16h a0 = frag_a(Ae, rbase + mt0 * 16, k0, lda);
    v16h a1 = frag_a(Ae, rbase + (mt0 + 2) * 16, k0, lda);
    c0 = wmma_f16(a0, b, c0);
    c1 = wmma_f16(a1, b, c1);
  }

  const int hl   = lane >> 4;
  const int ncol = nbase + nt * 16 + (lane & 15);
  const float bb = be[ncol];
#pragma unroll
  for (int v = 0; v < 8; ++v) {
#pragma unroll
    for (int piece = 0; piece < 2; ++piece) {
      int grow  = rbase + (piece == 0 ? mt0 : mt0 + 2) * 16 + v + 8 * hl;
      float val = (piece == 0 ? c0[v] : c1[v]) + bb;
      if (grow < M) {
        size_t idx = (size_t)e * OstrideE + (size_t)grow * OstrideRow + ncol;
        if (do_gelu) {
          val = 0.5f * val * (1.0f + erff(val * 0.70710678118654752f));
          outH[idx] = (_Float16)val;
        } else {
          outF[idx] = val;
        }
      }
    }
  }
}

// ---------------------------------------------------------------------------
// BatchNorm (training stats over N, biased var). In-place on H, and writes the
// transposed mm_out to d_out: out[n*1280 + o*5 + e].
// ---------------------------------------------------------------------------
__global__ __launch_bounds__(256)
void bn_kernel(float* __restrict__ H, const float* __restrict__ gamma,
               const float* __restrict__ beta, float* __restrict__ out) {
  const int col = blockIdx.x;      // 0..1279  (e*256+o)
  const int e = col >> 8, o = col & 255;
  const int t = threadIdx.x;
  const size_t base = (size_t)e * N_ENT * ST_DIM + o;
  float s = 0.f, s2 = 0.f;
  for (int n = t; n < N_ENT; n += 256) {
    float v = H[base + (size_t)n * ST_DIM];
    s += v; s2 += v * v;
  }
  __shared__ float r1[256], r2[256];
  r1[t] = s; r2[t] = s2;
  __syncthreads();
  for (int st = 128; st > 0; st >>= 1) {
    if (t < st) { r1[t] += r1[t + st]; r2[t] += r2[t + st]; }
    __syncthreads();
  }
  const float mu    = r1[0] * (1.0f / N_ENT);
  const float var   = r2[0] * (1.0f / N_ENT) - mu * mu;
  const float scale = gamma[col] * rsqrtf(var + BN_EPS);
  const float shift = beta[col] - mu * scale;
  for (int n = t; n < N_ENT; n += 256) {
    float v = H[base + (size_t)n * ST_DIM];
    float y = v * scale + shift;
    H[base + (size_t)n * ST_DIM] = y;
    out[(size_t)n * (ST_DIM * COMPS) + (size_t)o * COMPS + e] = y;
  }
}

// ---------------------------------------------------------------------------
// Row L2-normalize enc -> mn (f16, padded rows) and st_feats -> sn (f16, padded)
// ---------------------------------------------------------------------------
__global__ __launch_bounds__(256)
void rownorm_kernel(const float* __restrict__ enc, const float* __restrict__ st,
                    _Float16* __restrict__ mn_h, _Float16* __restrict__ sn_h) {
  const int n = blockIdx.x, c = blockIdx.y, t = threadIdx.x;
  float xm = enc[((size_t)c * N_ENT + n) * ST_DIM + t];
  float xs = st[(size_t)n * (ST_DIM * COMPS) + (size_t)t * COMPS + c];
  __shared__ float r1[256], r2[256];
  r1[t] = xm * xm; r2[t] = xs * xs;
  __syncthreads();
  for (int s = 128; s > 0; s >>= 1) {
    if (t < s) { r1[t] += r1[t + s]; r2[t] += r2[t + s]; }
    __syncthreads();
  }
  float im = 1.0f / (sqrtf(r1[0]) + 1e-12f);
  float is = 1.0f / (sqrtf(r2[0]) + 1e-12f);
  const size_t po = ((size_t)c * PADN + n) * ST_DIM + t;
  mn_h[po] = (_Float16)(xm * im);
  sn_h[po] = (_Float16)(xs * is);
}

// ---------------------------------------------------------------------------
// Pass A: streamed S = sn@mn^T. LDS-staged A (once) and B (per col tile).
// Per row: argmax_j s_ij and sum_j exp(s_ij/tau).
// ---------------------------------------------------------------------------
__global__ __launch_bounds__(256)
void sim_pass_a_kernel(const _Float16* __restrict__ sn, const _Float16* __restrict__ mn,
                       float* __restrict__ rowsum, int* __restrict__ rowarg) {
  const int c     = blockIdx.y;
  const int rbase = blockIdx.x * 64;
  const _Float16* SN = sn + (size_t)c * PADN * ST_DIM;
  const _Float16* MN = mn + (size_t)c * PADN * ST_DIM;
  const int t = threadIdx.x, w = t >> 5, lane = t & 31;
  const int nt = w & 3, mt0 = w >> 2, hl = lane >> 4;
  const int r = t & 63, ch = t >> 6;

  __shared__ _Float16 aT[64 * LDB];
  __shared__ _Float16 bT[64 * LDB];
  __shared__ float sT[64][65];

  stage_contig(aT, SN + (size_t)rbase * ST_DIM);

  float bestv = -3.0e38f; int besti = 0; float esum = 0.f;

  for (int jt = 0; jt < PADN; jt += 64) {
    stage_contig(bT, MN + (size_t)jt * ST_DIM);
    __syncthreads();
    v8f c0 = {}; v8f c1 = {};
#pragma unroll
    for (int k0 = 0; k0 < ST_DIM; k0 += 32) {
      v16h b  = frag_b(bT, nt * 16, k0, LDB);
      v16h a0 = frag_a(aT, mt0 * 16, k0, LDB);
      v16h a1 = frag_a(aT, (mt0 + 2) * 16, k0, LDB);
      c0 = wmma_f16(a0, b, c0);
      c1 = wmma_f16(a1, b, c1);
    }
#pragma unroll
    for (int v = 0; v < 8; ++v) {
      sT[mt0 * 16 + v + 8 * hl][nt * 16 + (lane & 15)]       = c0[v];
      sT[(mt0 + 2) * 16 + v + 8 * hl][nt * 16 + (lane & 15)] = c1[v];
    }
    __syncthreads();
#pragma unroll
    for (int jj = 0; jj < 16; ++jj) {
      int lc = ch * 16 + jj, j = jt + lc;
      if (j < N_ENT) {
        float s = sT[r][lc];
        esum += expf(s * INV_TAU);
        if (s > bestv) { bestv = s; besti = j; }
      }
    }
    __syncthreads();
  }

  __shared__ float mv[64][4]; __shared__ int mi[64][4]; __shared__ float ms[64][4];
  mv[r][ch] = bestv; mi[r][ch] = besti; ms[r][ch] = esum;
  __syncthreads();
  if (ch == 0) {
    float bv = mv[r][0]; int bi = mi[r][0]; float es = ms[r][0];
    for (int q = 1; q < 4; ++q) {
      es += ms[r][q];
      float v2 = mv[r][q]; int i2 = mi[r][q];
      if (v2 > bv || (v2 == bv && i2 < bi)) { bv = v2; bi = i2; }
    }
    int row = rbase + r;
    if (row < N_ENT) {
      rowsum[c * N_ENT + row] = es;
      rowarg[c * N_ENT + row] = bi;
    }
  }
}

// ---------------------------------------------------------------------------
// Pass B: rows = mn[argmax] @ mn^T streamed; per row top-6; then
// pos = sum_t exp(sn[i].mn[top_t]/tau); loss_i = log(rowsum_i) - log(pos).
// ---------------------------------------------------------------------------
__global__ __launch_bounds__(256)
void sim_pass_b_kernel(const _Float16* __restrict__ sn, const _Float16* __restrict__ mn,
                       const float* __restrict__ rowsum, const int* __restrict__ rowarg,
                       float* __restrict__ loss_row) {
  const int c     = blockIdx.y;
  const int rbase = blockIdx.x * 64;
  const _Float16* SN = sn + (size_t)c * PADN * ST_DIM;
  const _Float16* MN = mn + (size_t)c * PADN * ST_DIM;
  const int t = threadIdx.x, w = t >> 5, lane = t & 31;
  const int nt = w & 3, mt0 = w >> 2, hl = lane >> 4;
  const int r = t & 63, ch = t >> 6;

  __shared__ int gIdx[64];
  if (t < 64) {
    int row = rbase + t;
    gIdx[t] = (row < N_ENT) ? rowarg[c * N_ENT + row] : 0;
  }
  __syncthreads();

  __shared__ _Float16 aT[64 * LDB];
  __shared__ _Float16 bT[64 * LDB];
  __shared__ float sT[64][65];

  stage_gather(aT, MN, gIdx);

  float tv[TOPK]; int ti[TOPK];
#pragma unroll
  for (int q = 0; q < TOPK; ++q) { tv[q] = -3.0e38f; ti[q] = 0x7fffffff; }

  for (int jt = 0; jt < PADN; jt += 64) {
    stage_contig(bT, MN + (size_t)jt * ST_DIM);
    __syncthreads();
    v8f c0 = {}; v8f c1 = {};
#pragma unroll
    for (int k0 = 0; k0 < ST_DIM; k0 += 32) {
      v16h b  = frag_b(bT, nt * 16, k0, LDB);
      v16h a0 = frag_a(aT, mt0 * 16, k0, LDB);
      v16h a1 = frag_a(aT, (mt0 + 2) * 16, k0, LDB);
      c0 = wmma_f16(a0, b, c0);
      c1 = wmma_f16(a1, b, c1);
    }
#pragma unroll
    for (int v = 0; v < 8; ++v) {
      sT[mt0 * 16 + v + 8 * hl][nt * 16 + (lane & 15)]       = c0[v];
      sT[(mt0 + 2) * 16 + v + 8 * hl][nt * 16 + (lane & 15)] = c1[v];
    }
    __syncthreads();
#pragma unroll
    for (int jj = 0; jj < 16; ++jj) {
      int lc = ch * 16 + jj, j = jt + lc;
      if (j < N_ENT) {
        float s = sT[r][lc];
        if (s > tv[TOPK - 1]) {
          int p = TOPK - 1;
          while (p > 0 && s > tv[p - 1]) { tv[p] = tv[p - 1]; ti[p] = ti[p - 1]; --p; }
          tv[p] = s; ti[p] = j;
        }
      }
    }
    __syncthreads();
  }

  __shared__ float cv[64][4 * TOPK];
  __shared__ int   ci[64][4 * TOPK];
#pragma unroll
  for (int q = 0; q < TOPK; ++q) { cv[r][ch * TOPK + q] = tv[q]; ci[r][ch * TOPK + q] = ti[q]; }
  __syncthreads();

  if (ch == 0) {
    int row = rbase + r;
    if (row < N_ENT) {
      bool used[4 * TOPK];
      for (int q = 0; q < 4 * TOPK; ++q) used[q] = false;
      const _Float16* snr = SN + (size_t)row * ST_DIM;
      float pos = 0.f;
      for (int sel = 0; sel < TOPK; ++sel) {
        float bv = -3.4e38f; int bq = 0; int bidx = 0x7fffffff;
        for (int q = 0; q < 4 * TOPK; ++q) {
          if (used[q]) continue;
          float v2 = cv[r][q]; int i2 = ci[r][q];
          if (v2 > bv || (v2 == bv && i2 < bidx)) { bv = v2; bq = q; bidx = i2; }
        }
        used[bq] = true;
        if (bidx < N_ENT) {
          const _Float16* mnr = MN + (size_t)bidx * ST_DIM;
          float d = 0.f;
          for (int k = 0; k < ST_DIM; ++k) d += (float)snr[k] * (float)mnr[k];
          pos += expf(d * INV_TAU);
        }
      }
      loss_row[c * N_ENT + row] = logf(rowsum[c * N_ENT + row]) - logf(pos);
    }
  }
}

// ---------------------------------------------------------------------------
// Deterministic fixed-order final loss reduction.
// ---------------------------------------------------------------------------
__global__ __launch_bounds__(256)
void loss_kernel(const float* __restrict__ loss_row, float* __restrict__ out_loss) {
  const int t = threadIdx.x;
  float s = 0.f;
  for (int i = t; i < COMPS * N_ENT; i += 256) s += loss_row[i];
  __shared__ float red[256];
  red[t] = s;
  __syncthreads();
  for (int st = 128; st > 0; st >>= 1) {
    if (t < st) red[t] += red[t + st];
    __syncthreads();
  }
  if (t == 0) out_loss[0] = red[0] * (1.0f / (COMPS * N_ENT));
}

// ---------------------------------------------------------------------------
extern "C" void kernel_launch(void* const* d_in, const int* in_sizes, int n_in,
                              void* d_out, int out_size, void* d_ws, size_t ws_size,
                              hipStream_t stream) {
  const float* st    = (const float*)d_in[0];
  const float* mm    = (const float*)d_in[1];
  const float* W1    = (const float*)d_in[2];
  const float* b1    = (const float*)d_in[3];
  const float* W2    = (const float*)d_in[4];
  const float* b2    = (const float*)d_in[5];
  const float* gamma = (const float*)d_in[6];
  const float* beta  = (const float*)d_in[7];
  float* out = (float*)d_out;

  char* ws = (char*)d_ws;
  size_t off = 0;
  auto alloc = [&](size_t bytes) -> void* {
    void* p = ws + off;
    off += (bytes + 255) & ~(size_t)255;
    return p;
  };
  _Float16* mm_h = (_Float16*)alloc((size_t)PADN * MM_DIM * 2);
  _Float16* W1h  = (_Float16*)alloc((size_t)COMPS * ST_DIM * MM_DIM * 2);
  _Float16* W2h  = (_Float16*)alloc((size_t)COMPS * ST_DIM * ST_DIM * 2);
  _Float16* H1h  = (_Float16*)alloc((size_t)COMPS * PADN * ST_DIM * 2);
  float*    H2   = (float*)   alloc((size_t)COMPS * N_ENT * ST_DIM * 4);
  _Float16* mn_h = (_Float16*)alloc((size_t)COMPS * PADN * ST_DIM * 2);
  _Float16* sn_h = (_Float16*)alloc((size_t)COMPS * PADN * ST_DIM * 2);
  float*    rsum = (float*)   alloc((size_t)COMPS * N_ENT * 4);
  int*      rarg = (int*)     alloc((size_t)COMPS * N_ENT * 4);
  float*    lrow = (float*)   alloc((size_t)COMPS * N_ENT * 4);

  const int PAD = PADN - N_ENT;   // 48 rows

  // Casts
  int nc;
  nc = N_ENT * MM_DIM;
  f32_to_f16_kernel<<<(nc + 255) / 256, 256, 0, stream>>>(mm, mm_h, nc);
  nc = COMPS * ST_DIM * MM_DIM;
  f32_to_f16_kernel<<<(nc + 255) / 256, 256, 0, stream>>>(W1, W1h, nc);
  nc = COMPS * ST_DIM * ST_DIM;
  f32_to_f16_kernel<<<(nc + 255) / 256, 256, 0, stream>>>(W2, W2h, nc);

  // Zero padding rows so WMMA fragment loads need no bounds checks.
  nc = PAD * MM_DIM;
  zero_f16_kernel<<<(nc + 255) / 256, 256, 0, stream>>>(mm_h + (size_t)N_ENT * MM_DIM, nc);
  nc = PAD * ST_DIM;
  for (int e = 0; e < COMPS; ++e) {
    size_t po = ((size_t)e * PADN + N_ENT) * ST_DIM;
    zero_f16_kernel<<<(nc + 255) / 256, 256, 0, stream>>>(H1h + po, nc);
    zero_f16_kernel<<<(nc + 255) / 256, 256, 0, stream>>>(mn_h + po, nc);
    zero_f16_kernel<<<(nc + 255) / 256, 256, 0, stream>>>(sn_h + po, nc);
  }

  dim3 g1(PADN / 64, ST_DIM / 64, COMPS);
  // GEMM1: H1 = GELU(mm @ W1^T + b1)  -> f16 (padded rows)
  enc_gemm_kernel<<<g1, 256, 0, stream>>>(mm_h, 0, MM_DIM,
                                          W1h, (size_t)ST_DIM * MM_DIM, MM_DIM,
                                          b1, H1h, nullptr,
                                          (size_t)PADN * ST_DIM, ST_DIM, N_ENT, MM_DIM, 1);
  // GEMM2: H2 = H1 @ W2^T + b2 -> f32 (unpadded rows)
  enc_gemm_kernel<<<g1, 256, 0, stream>>>(H1h, (size_t)PADN * ST_DIM, ST_DIM,
                                          W2h, (size_t)ST_DIM * ST_DIM, ST_DIM,
                                          b2, nullptr, H2,
                                          (size_t)N_ENT * ST_DIM, ST_DIM, N_ENT, ST_DIM, 0);

  bn_kernel<<<COMPS * ST_DIM, 256, 0, stream>>>(H2, gamma, beta, out);

  rownorm_kernel<<<dim3(N_ENT, COMPS), 256, 0, stream>>>(H2, st, mn_h, sn_h);

  dim3 gs(PADN / 64, COMPS);
  sim_pass_a_kernel<<<gs, 256, 0, stream>>>(sn_h, mn_h, rsum, rarg);
  sim_pass_b_kernel<<<gs, 256, 0, stream>>>(sn_h, mn_h, rsum, rarg, lrow);

  loss_kernel<<<1, 256, 0, stream>>>(lrow, out + (size_t)N_ENT * ST_DIM * COMPS);

  (void)in_sizes; (void)n_in; (void)out_size; (void)ws_size;
}
